// CollaborationGNN_39548058862204
// MI455X (gfx1250) — matrix operand order repaired and checked by reading, not compiled, not run on hardware
//
#include <hip/hip_runtime.h>
#include <hip/hip_bf16.h>

// ---------------------------------------------------------------------------
// CDNA5 (gfx1250) fused GNN link-prediction pipeline:
//   v_wmma_f32_16x16x32_f16 GEMMs + global_load_async_to_lds_b128 gather
// ---------------------------------------------------------------------------

typedef __attribute__((ext_vector_type(16))) _Float16 v16h;
typedef __attribute__((ext_vector_type(8)))  _Float16 v8h;
typedef __attribute__((ext_vector_type(8)))  float    v8f;

#define IN_C  128
#define HID   256
#define OUT_C 128
#define EPSV  1e-5f

// ---- WMMA fragment helpers (layouts per CDNA5 ISA 7.12.2, wave32) ----------
// A (16x32 f16, MxK): lane l holds row m=l&15; halves j=0..7 -> K = base+j,
// j=8..15 -> K = 16+base+j-8, with base = (l>=16)?8:0.
__device__ __forceinline__ v16h load_a_frag(const _Float16* row0, int strideH,
                                            int k0, int lane) {
  const int m = lane & 15;
  const int b = (lane >> 4) << 3;
  const _Float16* p = row0 + (long)m * strideH + k0 + b;
  union { v16h v; struct { v8h lo; v8h hi; } s; } u;
  u.s.lo = *(const v8h*)(p);
  u.s.hi = *(const v8h*)(p + 16);
  return u.v;
}

// B (32x16 f16, KxN), stored as W rows (out = q @ W^T so B column n = W row n):
// lane l holds column n=l&15; j=0..7 -> K = base+j, j=8..15 -> K = 16+base+j-8, base = (l>=16)?8:0.
__device__ __forceinline__ v16h load_b_frag(const _Float16* w, int strideH,
                                            int n0, int k0, int lane) {
  const int n  = n0 + (lane & 15);
  const int kk = k0 + ((lane >> 4) << 3);
  const _Float16* p = w + (long)n * strideH + kk;
  union { v16h v; struct { v8h lo; v8h hi; } s; } u;
  u.s.lo = *(const v8h*)(p);
  u.s.hi = *(const v8h*)(p + 16);
  return u.v;
}

__device__ __forceinline__ v8f wmma32(v16h a, v16h b, v8f c) {
  return __builtin_amdgcn_wmma_f32_16x16x32_f16(false, a, false, b,
                                                (short)0, c, false, false);
}

// ---------------------------------------------------------------------------
// prep: fold BN params (bias folded in), build f16 weights.
// Wcat1[n] = [W1l[n,:] , W1r[n,:]]  (256 x 256)   -> SAGE1 single GEMM, K=256
// Wcat2[n] = [W2l[n,:] , W2r[n,:]]  (128 x 512)   -> SAGE2 single GEMM, K=512
// ---------------------------------------------------------------------------
__global__ void prep_kernel(const float* __restrict__ W1l, const float* __restrict__ W1r,
                            const float* __restrict__ b1,  const float* __restrict__ g1,
                            const float* __restrict__ bb1, const float* __restrict__ m1,
                            const float* __restrict__ v1,
                            const float* __restrict__ W2l, const float* __restrict__ W2r,
                            const float* __restrict__ b2,
                            const float* __restrict__ pW1, const float* __restrict__ pb1,
                            const float* __restrict__ pg1, const float* __restrict__ pbb1,
                            const float* __restrict__ pm1, const float* __restrict__ pv1,
                            const float* __restrict__ pW2, const float* __restrict__ pb2,
                            const float* __restrict__ pg2, const float* __restrict__ pbb2,
                            const float* __restrict__ pm2, const float* __restrict__ pv2,
                            _Float16* __restrict__ Wcat1, _Float16* __restrict__ Wcat2,
                            _Float16* __restrict__ pW1h,  _Float16* __restrict__ pW2h,
                            float* __restrict__ s1, float* __restrict__ t1,
                            float* __restrict__ t2f,
                            float* __restrict__ ps1, float* __restrict__ pt1,
                            float* __restrict__ ps2, float* __restrict__ pt2) {
  const int tid = blockIdx.x * blockDim.x + threadIdx.x;
  const int st  = gridDim.x * blockDim.x;
  for (int i = tid; i < 256 * 256; i += st) {
    int n = i >> 8, k = i & 255;
    float v = (k < 128) ? W1l[n * 128 + k] : W1r[n * 128 + (k - 128)];
    Wcat1[i] = (_Float16)v;
  }
  for (int i = tid; i < 128 * 512; i += st) {
    int n = i >> 9, k = i & 511;
    float v = (k < 256) ? W2l[n * 256 + k] : W2r[n * 256 + (k - 256)];
    Wcat2[i] = (_Float16)v;
  }
  for (int i = tid; i < 256 * 256; i += st) pW1h[i] = (_Float16)pW1[i];
  for (int i = tid; i < 128 * 256; i += st) pW2h[i] = (_Float16)pW2[i];
  for (int i = tid; i < 256; i += st) {
    float sc = g1[i] * rsqrtf(v1[i] + EPSV);
    s1[i] = sc;
    t1[i] = (b1[i] - m1[i]) * sc + bb1[i];
    float psc = pg1[i] * rsqrtf(pv1[i] + EPSV);
    ps1[i] = psc;
    pt1[i] = (pb1[i] - pm1[i]) * psc + pbb1[i];
  }
  for (int i = tid; i < 128; i += st) {
    t2f[i] = b2[i];
    float psc = pg2[i] * rsqrtf(pv2[i] + EPSV);
    ps2[i] = psc;
    pt2[i] = (pb2[i] - pm2[i]) * psc + pbb2[i];
  }
}

__global__ void zero_kernel(float* __restrict__ p, long n) {
  long i  = (long)blockIdx.x * blockDim.x + threadIdx.x;
  long st = (long)gridDim.x * blockDim.x;
  for (; i < n; i += st) p[i] = 0.0f;
}

// ---------------------------------------------------------------------------
// scatter: one wave per edge (grid-strided); f32 atomics hit L2-resident agg.
// ---------------------------------------------------------------------------
__global__ void scatter_kernel(const float* __restrict__ feat,
                               const int* __restrict__ src, const int* __restrict__ dst,
                               float* __restrict__ agg, float* __restrict__ cnt,
                               int E, int F) {
  const int lane = threadIdx.x & 31;
  const int wid  = (blockIdx.x * blockDim.x + threadIdx.x) >> 5;
  const int nw   = (gridDim.x * blockDim.x) >> 5;
  for (int e = wid; e < E; e += nw) {
    const int s = src[e], d = dst[e];
    const float* fr = feat + (long)s * F;
    float* ar = agg + (long)d * F;
    for (int k = lane; k < F; k += 32) atomicAdd(&ar[k], fr[k]);
    if (cnt != nullptr && lane == 0) atomicAdd(&cnt[d], 1.0f);
  }
}

// ---------------------------------------------------------------------------
// SAGE GEMM: per-wave 16-node tile. A row = [agg/max(cnt,1) , self] (K = 2F),
// staged in LDS as f16; GEMM vs Wcat (NOUT x K) via WMMA; fused scale/shift/ReLU.
// ---------------------------------------------------------------------------
template <int F, int NOUT>
__global__ __launch_bounds__(64) void sage_gemm_kernel(
    const float* __restrict__ agg, const float* __restrict__ cnt,
    const float* __restrict__ self, const _Float16* __restrict__ Wh,
    const float* __restrict__ scale, const float* __restrict__ shift,
    int relu, float* __restrict__ outF, _Float16* __restrict__ outH,
    int nTiles, int Nn) {
  constexpr int K = 2 * F;
  __shared__ __align__(16) _Float16 Ash[2 * 16 * K];
  const int lane = threadIdx.x & 31;
  const int wv   = threadIdx.x >> 5;
  const int tile = blockIdx.x * 2 + wv;
  if (tile >= nTiles) return;
  _Float16* A = Ash + wv * 16 * K;
  const int node0 = tile * 16;

  for (int r = 0; r < 16; ++r) {
    const int node = min(node0 + r, Nn - 1);
    const float rc = 1.0f / fmaxf(cnt[node], 1.0f);
    const float* ag = agg  + (long)node * F;
    const float* sf = self + (long)node * F;
    for (int k = lane; k < F; k += 32) {
      A[r * K + k]     = (_Float16)(ag[k] * rc);
      A[r * K + F + k] = (_Float16)(sf[k]);
    }
  }

  for (int nt = 0; nt < NOUT / 16; ++nt) {
    v8f c = {};
#pragma unroll
    for (int kt = 0; kt < K / 32; ++kt) {
      v16h a = load_a_frag(A, K, kt * 32, lane);
      v16h b = load_b_frag(Wh, K, nt * 16, kt * 32, lane);
      c = wmma32(a, b, c);
    }
    const int n  = nt * 16 + (lane & 15);
    const float sc = scale ? scale[n] : 1.0f;
    const float sh = shift ? shift[n] : 0.0f;
    const int mb = (lane >> 4) << 3;
#pragma unroll
    for (int r = 0; r < 8; ++r) {
      const int row = node0 + mb + r;
      if (row < Nn) {
        float v = c[r] * sc + sh;
        if (relu) v = fmaxf(v, 0.0f);
        const long o = (long)row * NOUT + n;
        if (outF) outF[o] = v;
        if (outH) outH[o] = (_Float16)v;
      }
    }
  }
}

// ---------------------------------------------------------------------------
// Fused link decoder: one wave per 32-edge tile, fully in LDS.
//   gather: global_load_async_to_lds_b128 (ASYNCcnt), z is f16 + L2-resident
//   L1: 256->256, B-frag reused across 2 M-tiles (c0/c1); BN+ReLU -> q2
//   L2: 256->128, same; BN+ReLU -> q1 (reused as 32x128)
//   L3: 128->1 per-lane dot, lane-staggered K to dodge LDS bank conflicts
// ---------------------------------------------------------------------------
__global__ __launch_bounds__(32) void decoder_kernel(
    const int* __restrict__ pa, const int* __restrict__ pb,
    const _Float16* __restrict__ zh,
    const _Float16* __restrict__ W1h,
    const float* __restrict__ s1, const float* __restrict__ t1,
    const _Float16* __restrict__ W2h,
    const float* __restrict__ s2, const float* __restrict__ t2,
    const float* __restrict__ w3, const float* __restrict__ b3,
    float* __restrict__ out, int nTiles, int PEn) {
  __shared__ __align__(16) _Float16 sm[32 * 512];  // q1 32x256 | q2 32x256
  const int lane = threadIdx.x & 31;
  const int tile = blockIdx.x;
  if (tile >= nTiles) return;
  _Float16* q1 = sm;
  _Float16* q2 = sm + 32 * 256;
  const int e0 = tile * 32;

  // async gather: lane<16 copies 16B chunk of z[a], lane>=16 of z[b]
  {
    const int half  = lane >> 4;
    const int chunk = lane & 15;
    for (int r = 0; r < 32; ++r) {
      const int e = min(e0 + r, PEn - 1);
      const int node = half ? pb[e] : pa[e];
      const unsigned long long gaddr =
          (unsigned long long)(const void*)((const char*)(zh + (long)node * 128) + chunk * 16);
      const unsigned ldsa =
          (unsigned)(size_t)(const void*)(q1 + r * 256 + half * 128 + chunk * 8);
      asm volatile("global_load_async_to_lds_b128 %0, %1, off"
                   :: "v"(ldsa), "v"(gaddr) : "memory");
    }
    asm volatile("s_wait_asynccnt 0x0" ::: "memory");
  }

  // layer 1: 256 -> 256
  for (int nt = 0; nt < 16; ++nt) {
    v8f c0 = {}, c1 = {};
#pragma unroll
    for (int kt = 0; kt < 8; ++kt) {
      v16h b  = load_b_frag(W1h, 256, nt * 16, kt * 32, lane);
      v16h a0 = load_a_frag(q1, 256, kt * 32, lane);
      v16h a1 = load_a_frag(q1 + 16 * 256, 256, kt * 32, lane);
      c0 = wmma32(a0, b, c0);
      c1 = wmma32(a1, b, c1);
    }
    const int n  = nt * 16 + (lane & 15);
    const float sc = s1[n], sh = t1[n];
    const int mb = (lane >> 4) << 3;
#pragma unroll
    for (int r = 0; r < 8; ++r) {
      q2[(mb + r) * 256 + n]      = (_Float16)fmaxf(c0[r] * sc + sh, 0.0f);
      q2[(16 + mb + r) * 256 + n] = (_Float16)fmaxf(c1[r] * sc + sh, 0.0f);
    }
  }

  // layer 2: 256 -> 128 (writes q1, reused as 32x128)
  for (int nt = 0; nt < 8; ++nt) {
    v8f c0 = {}, c1 = {};
#pragma unroll
    for (int kt = 0; kt < 8; ++kt) {
      v16h b  = load_b_frag(W2h, 256, nt * 16, kt * 32, lane);
      v16h a0 = load_a_frag(q2, 256, kt * 32, lane);
      v16h a1 = load_a_frag(q2 + 16 * 256, 256, kt * 32, lane);
      c0 = wmma32(a0, b, c0);
      c1 = wmma32(a1, b, c1);
    }
    const int n  = nt * 16 + (lane & 15);
    const float sc = s2[n], sh = t2[n];
    const int mb = (lane >> 4) << 3;
#pragma unroll
    for (int r = 0; r < 8; ++r) {
      q1[(mb + r) * 128 + n]      = (_Float16)fmaxf(c0[r] * sc + sh, 0.0f);
      q1[(16 + mb + r) * 128 + n] = (_Float16)fmaxf(c1[r] * sc + sh, 0.0f);
    }
  }

  // layer 3: 128 -> 1 ; one row per lane, staggered K order (bank spread)
  {
    float acc = 0.0f;
    const _Float16* q3 = q1 + lane * 128;
#pragma unroll 16
    for (int i = 0; i < 128; ++i) {
      const int k = (i + lane * 4) & 127;
      acc += (float)q3[k] * w3[k];
    }
    if ((e0 + lane) < PEn) out[e0 + lane] = acc + b3[0];
  }
}

// ---------------------------------------------------------------------------
extern "C" void kernel_launch(void* const* d_in, const int* in_sizes, int n_in,
                              void* d_out, int out_size, void* d_ws, size_t ws_size,
                              hipStream_t stream) {
  (void)n_in; (void)out_size; (void)ws_size;
  const float* x   = (const float*)d_in[0];
  const int*   ei  = (const int*)d_in[1];
  const int*   pei = (const int*)d_in[2];
  const float* W1l = (const float*)d_in[3];
  const float* W1r = (const float*)d_in[4];
  const float* b1  = (const float*)d_in[5];
  const float* g1  = (const float*)d_in[6];
  const float* bb1 = (const float*)d_in[7];
  const float* m1  = (const float*)d_in[8];
  const float* v1  = (const float*)d_in[9];
  const float* W2l = (const float*)d_in[10];
  const float* W2r = (const float*)d_in[11];
  const float* b2  = (const float*)d_in[12];
  const float* pW1 = (const float*)d_in[13];
  const float* pb1 = (const float*)d_in[14];
  const float* pg1 = (const float*)d_in[15];
  const float* pbb1= (const float*)d_in[16];
  const float* pm1 = (const float*)d_in[17];
  const float* pv1 = (const float*)d_in[18];
  const float* pW2 = (const float*)d_in[19];
  const float* pb2 = (const float*)d_in[20];
  const float* pg2 = (const float*)d_in[21];
  const float* pbb2= (const float*)d_in[22];
  const float* pm2 = (const float*)d_in[23];
  const float* pv2 = (const float*)d_in[24];
  const float* pW3 = (const float*)d_in[25];
  const float* pb3 = (const float*)d_in[26];

  const int N  = in_sizes[0] / IN_C;  // 50000
  const int E  = in_sizes[1] / 2;     // 640000
  const int PE = in_sizes[2] / 2;     // 500000
  const int* src = ei;
  const int* dst = ei + E;
  const int* pa  = pei;
  const int* pb  = pei + PE;

  char* ws = (char*)d_ws;
  size_t off = 0;
  auto alloc = [&](size_t bytes) {
    void* p = ws + off;
    off = (off + bytes + 255) & ~(size_t)255;
    return p;
  };
  float*    aggBuf = (float*)alloc((size_t)N * HID * 4);      // agg1 then agg2
  float*    cnt    = (float*)alloc((size_t)N * 4);            // in-degree (shared)
  float*    h      = (float*)alloc((size_t)N * HID * 4);      // layer-1 act f32
  _Float16* zh     = (_Float16*)alloc((size_t)N * OUT_C * 2); // z f16 (L2-resident)
  _Float16* Wcat1  = (_Float16*)alloc(256 * 256 * 2);
  _Float16* Wcat2  = (_Float16*)alloc(128 * 512 * 2);
  _Float16* pW1h   = (_Float16*)alloc(256 * 256 * 2);
  _Float16* pW2h   = (_Float16*)alloc(128 * 256 * 2);
  float* s1  = (float*)alloc(256 * 4);
  float* t1  = (float*)alloc(256 * 4);
  float* t2f = (float*)alloc(128 * 4);
  float* ps1 = (float*)alloc(256 * 4);
  float* pt1 = (float*)alloc(256 * 4);
  float* ps2 = (float*)alloc(128 * 4);
  float* pt2 = (float*)alloc(128 * 4);

  prep_kernel<<<128, 256, 0, stream>>>(W1l, W1r, b1, g1, bb1, m1, v1,
                                       W2l, W2r, b2,
                                       pW1, pb1, pg1, pbb1, pm1, pv1,
                                       pW2, pb2, pg2, pbb2, pm2, pv2,
                                       Wcat1, Wcat2, pW1h, pW2h,
                                       s1, t1, t2f, ps1, pt1, ps2, pt2);

  zero_kernel<<<2048, 256, 0, stream>>>(aggBuf, (long)N * IN_C);
  zero_kernel<<<256, 256, 0, stream>>>(cnt, (long)N);
  scatter_kernel<<<2048, 256, 0, stream>>>(x, src, dst, aggBuf, cnt, E, IN_C);

  const int nodeTiles = (N + 15) / 16;
  sage_gemm_kernel<IN_C, HID><<<(nodeTiles + 1) / 2, 64, 0, stream>>>(
      aggBuf, cnt, x, Wcat1, s1, t1, /*relu=*/1, h, (_Float16*)nullptr, nodeTiles, N);

  zero_kernel<<<2048, 256, 0, stream>>>(aggBuf, (long)N * HID);
  scatter_kernel<<<2048, 256, 0, stream>>>(h, src, dst, aggBuf, (float*)nullptr, E, HID);

  sage_gemm_kernel<HID, OUT_C><<<(nodeTiles + 1) / 2, 64, 0, stream>>>(
      aggBuf, cnt, h, Wcat2, (const float*)nullptr, t2f, /*relu=*/0,
      (float*)nullptr, zh, nodeTiles, N);

  const int edgeTiles = (PE + 31) / 32;
  decoder_kernel<<<edgeTiles, 32, 0, stream>>>(
      pa, pb, zh, pW1h, ps1, pt1, pW2h, ps2, pt2, pW3, pb3,
      (float*)d_out, edgeTiles, PE);
}
